// PairStackBlock_71116068487706
// MI455X (gfx1250) — compile-verified, matrix operand order checked by reading, not run on hardware
//
#include <hip/hip_runtime.h>

// ---------------------------------------------------------------------------
// MI455X (gfx1250) implementation of an AlphaFold-style PairStack block.
// wave32, WMMA bf16 16x16x32, fused epilogues, LDS-staged operands,
// async global->LDS staging, software-pipelined WMMA loops.
// ---------------------------------------------------------------------------

typedef __attribute__((ext_vector_type(16))) __bf16 v16bf;
typedef __attribute__((ext_vector_type(8)))  __bf16 v8bf;
typedef __attribute__((ext_vector_type(8)))  float  v8f;

#define DEV __device__ __forceinline__

// A-style WMMA fragment load (also valid for B when source is [N,K] K-contig):
// lane L (0..31): row = L&15, K-chunks at {0..7,16..23} (L<16) / {8..15,24..31}.
DEV v16bf load_tile(const __bf16* p, int ld) {
  int lane = threadIdx.x & 31;
  const __bf16* q = p + (lane & 15) * ld + ((lane >> 4) << 3);
  v8bf lo = *(const v8bf*)q;
  v8bf hi = *(const v8bf*)(q + 16);
  v16bf r;
#pragma unroll
  for (int i = 0; i < 8; ++i) { r[i] = lo[i]; r[i + 8] = hi[i]; }
  return r;
}

DEV v8f wmma_bf16(v16bf a, v16bf b, v8f c) {
  return __builtin_amdgcn_wmma_f32_16x16x32_bf16(false, a, false, b,
                                                 (short)0, c, false, false);
}

// gfx1250 async copy: 16 bytes global -> LDS, tracked by ASYNCcnt.
DEV void async_g2l_b128(unsigned lds_byte_addr, const void* gsrc) {
  asm volatile("global_load_async_to_lds_b128 %0, %1, off"
               :: "v"(lds_byte_addr), "v"(gsrc) : "memory");
}
DEV void wait_asynccnt0() {
  asm volatile("s_wait_asynccnt 0x0" ::: "memory");
}

// ---------------------------------------------------------------------------
// LayerNorm over C=128, one wave per token, bf16 output.
// ---------------------------------------------------------------------------
__global__ __launch_bounds__(256) void ln128_kernel(
    const float* __restrict__ x, const float* __restrict__ w,
    const float* __restrict__ b, __bf16* __restrict__ out, int T) {
  int t = blockIdx.x * 8 + (threadIdx.x >> 5);
  if (t >= T) return;
  int lane = threadIdx.x & 31;
  const float4 v = *(const float4*)(x + (long)t * 128 + lane * 4);
  float s = v.x + v.y + v.z + v.w;
  float q = v.x * v.x + v.y * v.y + v.z * v.z + v.w * v.w;
#pragma unroll
  for (int m = 1; m < 32; m <<= 1) {
    s += __shfl_xor(s, m);
    q += __shfl_xor(q, m);
  }
  float mu  = s * (1.0f / 128.0f);
  float inv = rsqrtf(q * (1.0f / 128.0f) - mu * mu + 1e-5f);
  float vv[4] = {v.x, v.y, v.z, v.w};
#pragma unroll
  for (int c = 0; c < 4; ++c) {
    int ci = lane * 4 + c;
    out[(long)t * 128 + ci] = (__bf16)((vv[c] - mu) * inv * w[ci] + b[ci]);
  }
}

// ---------------------------------------------------------------------------
// Generic WMMA GEMM with fused epilogue.
//   out[M,N] = epi( alpha * A[M,K](bf16) x W[N,K]^T + bias )
// Block = 256 threads (8 waves); block tile = 128(M) x 16(N); W tile in LDS.
// bf16 weights are staged with gfx1250 async global->LDS copies; f32 weights
// go through VGPRs for the f32->bf16 convert. Main loop is register
// double-buffered so next fragment loads overlap the current WMMA.
// Batched mode (gridDim.z>1) for the triangle einsum: A/W advance by batch
// stride, output element index = (row*outLd + col)*outStride + z.
// ---------------------------------------------------------------------------
enum {
  F_SIG = 1, F_RELU = 2, F_AUXMUL = 4, F_MASKMUL = 8, F_RES = 16,
  F_OUT_BF16 = 32, F_W_F32 = 64
};

__global__ __launch_bounds__(256) void gemm_epi_kernel(
    const __bf16* __restrict__ A, const void* __restrict__ W,
    const float* __restrict__ bias, const float* __restrict__ aux,
    const float* __restrict__ maskv, const float* __restrict__ res,
    void* __restrict__ out, int M, int N, int K, float alpha, int flags,
    long aBatch, long wBatch, int outLd, int outStride) {
  __shared__ __attribute__((aligned(16))) __bf16 wtile[16 * 512];
  const int tid = threadIdx.x, wave = tid >> 5, lane = tid & 31;
  const int nBase = blockIdx.y * 16;
  const int mBase = blockIdx.x * 128 + wave * 16;
  const int z = blockIdx.z;
  const __bf16* Ab = A + (long)z * aBatch;

  // Stage (and convert) the 16xK weight tile into LDS.
  if (flags & F_W_F32) {
    const float* Wf = (const float*)W + (long)z * wBatch;
    for (int idx = tid; idx < 16 * K; idx += 256) {
      int r = idx / K, c = idx - r * K;
      wtile[idx] = (__bf16)Wf[(long)(nBase + r) * K + c];
    }
  } else {
    // gfx1250 async path: 16B chunks, no VGPR round trip (ASYNCcnt).
    const __bf16* Wb = (const __bf16*)W + (long)z * wBatch;
    for (int idx = tid * 8; idx < 16 * K; idx += 256 * 8) {
      int r = idx / K, c = idx - r * K;
      async_g2l_b128((unsigned)(size_t)(wtile + idx),
                     Wb + (long)(nBase + r) * K + c);
    }
    wait_asynccnt0();
  }
  __syncthreads();

  v8f acc = {};
  const __bf16* ap = Ab + (long)mBase * K;
  // Software pipeline: fragment loads for step k+1 issue before WMMA of step k.
  v16bf a = load_tile(ap, K);
  v16bf b = load_tile(wtile, K);
  for (int k0 = 32; k0 < K; k0 += 32) {
    __builtin_prefetch(ap + k0 + 256, 0, 1);  // global_prefetch_b8
    v16bf an = load_tile(ap + k0, K);
    v16bf bn = load_tile(wtile + k0, K);
    acc = wmma_bf16(a, b, acc);
    a = an;
    b = bn;
  }
  acc = wmma_bf16(a, b, acc);

  const int row0 = mBase + ((lane >> 4) << 3);
  const int col  = nBase + (lane & 15);
  const float bv = bias ? bias[col] : 0.0f;
#pragma unroll
  for (int g = 0; g < 8; ++g) {
    long row = row0 + g;
    float val = acc[g] * alpha + bv;
    if (flags & F_SIG)     val = 1.0f / (1.0f + __expf(-val));
    if (flags & F_RELU)    val = fmaxf(val, 0.0f);
    if (flags & F_AUXMUL)  val *= aux[row * N + col];
    if (flags & F_MASKMUL) val *= maskv[row];
    if (flags & F_RES)     val += res[row * N + col];
    long oi = (row * (long)outLd + col) * (long)outStride + z;
    if (flags & F_OUT_BF16) ((__bf16*)out)[oi] = (__bf16)val;
    else                    ((float*)out)[oi] = val;
  }
}

// ---------------------------------------------------------------------------
// Triangle bias: tb[h][t] = sum_c zl[t,c] * b_w[h,c]  (H=4, C=128)
// ---------------------------------------------------------------------------
__global__ __launch_bounds__(256) void bias_tb_kernel(
    const __bf16* __restrict__ zl, const float* __restrict__ bw,
    float* __restrict__ tb, int T) {
  int t = blockIdx.x * 8 + (threadIdx.x >> 5);
  if (t >= T) return;
  int lane = threadIdx.x & 31;
  float xv[4];
#pragma unroll
  for (int c = 0; c < 4; ++c) xv[c] = (float)zl[(long)t * 128 + lane * 4 + c];
#pragma unroll
  for (int h = 0; h < 4; ++h) {
    float s = 0.f;
#pragma unroll
    for (int c = 0; c < 4; ++c) s += xv[c] * bw[h * 128 + lane * 4 + c];
#pragma unroll
    for (int m = 1; m < 32; m <<= 1) s += __shfl_xor(s, m);
    if (lane == 0) tb[(long)h * T + t] = s;
  }
}

// ---------------------------------------------------------------------------
// Triangle attention for one (row i, head h): softmax(QK^T/sqrt(c)+biases)V,
// gated by sigmoid(g). Q,K,V bf16 [t, H*32]; probs transposed through LDS.
// Dynamic LDS: V^T 16KB + per-wave prob tiles 64KB = 80KB (of 320KB/WGP).
// ---------------------------------------------------------------------------
__global__ __launch_bounds__(256) void attn_kernel(
    const __bf16* __restrict__ q, const __bf16* __restrict__ k,
    const __bf16* __restrict__ v, const float* __restrict__ gsig,
    const float* __restrict__ tb, const float* __restrict__ maskv,
    __bf16* __restrict__ obuf) {
  extern __shared__ char smem[];
  __bf16* vT    = (__bf16*)smem;                     // [32][256]
  __bf16* probs = (__bf16*)(smem + 32 * 256 * 2);    // [8 waves][16][256]
  const int i = blockIdx.x, h = blockIdx.y;
  const int tid = threadIdx.x, wave = tid >> 5, lane = tid & 31;
  const long base = ((long)i * 256) * 128 + h * 32;

  for (int idx = tid; idx < 256 * 32; idx += 256) {
    int kk = idx >> 5, c = idx & 31;
    vT[c * 256 + kk] = v[base + (long)kk * 128 + c];
  }
  __syncthreads();

  __bf16* pw = probs + wave * 16 * 256;
  const int rsub = (lane >> 4) << 3;
  const int col  = lane & 15;

  for (int jt = wave; jt < 16; jt += 8) {
    const int j0 = jt * 16;
    v16bf aQ = load_tile(q + base + (long)j0 * 128, 128);
    v8f acc[16];
#pragma unroll
    for (int kt = 0; kt < 16; ++kt) {
      v16bf bK = load_tile(k + base + (long)(kt * 16) * 128, 128);
      v8f zero = {};
      acc[kt] = wmma_bf16(aQ, bK, zero);
    }
    // Softmax over K (row = j0 + rsub + g, cols striped over 16 lanes x 16 tiles)
#pragma unroll
    for (int g = 0; g < 8; ++g) {
      const int jq = j0 + rsub + g;
      float vals[16];
      float mx = -3.0e38f;
#pragma unroll
      for (int kt = 0; kt < 16; ++kt) {
        int kres = kt * 16 + col;
        float s = acc[kt][g] + tb[(long)h * 65536 + (long)jq * 256 + kres] +
                  1.0e9f * (maskv[(long)i * 256 + kres] - 1.0f);
        vals[kt] = s;
        mx = fmaxf(mx, s);
      }
#pragma unroll
      for (int m = 1; m < 16; m <<= 1) mx = fmaxf(mx, __shfl_xor(mx, m));
      float sum = 0.f;
#pragma unroll
      for (int kt = 0; kt < 16; ++kt) {
        vals[kt] = __expf(vals[kt] - mx);
        sum += vals[kt];
      }
#pragma unroll
      for (int m = 1; m < 16; m <<= 1) sum += __shfl_xor(sum, m);
      float inv = 1.0f / sum;
#pragma unroll
      for (int kt = 0; kt < 16; ++kt)
        pw[(rsub + g) * 256 + kt * 16 + col] = (__bf16)(vals[kt] * inv);
    }
    // o = P x V (K=256 in 8 WMMA steps, double-buffered), gate, store bf16.
#pragma unroll
    for (int ct = 0; ct < 2; ++ct) {
      v8f accO = {};
      v16bf ap = load_tile(pw, 256);
      v16bf bv = load_tile(vT + ct * 16 * 256, 256);
#pragma unroll
      for (int k0 = 32; k0 < 256; k0 += 32) {
        v16bf apn = load_tile(pw + k0, 256);
        v16bf bvn = load_tile(vT + ct * 16 * 256 + k0, 256);
        accO = wmma_bf16(ap, bv, accO);
        ap = apn;
        bv = bvn;
      }
      accO = wmma_bf16(ap, bv, accO);
#pragma unroll
      for (int g = 0; g < 8; ++g) {
        int jq = j0 + rsub + g;
        int c  = ct * 16 + col;
        long oidx = ((long)i * 256 + jq) * 128 + h * 32 + c;
        obuf[oidx] = (__bf16)(accO[g] * gsig[oidx]);
      }
    }
  }
}

// ---------------------------------------------------------------------------
// [256,256,C] transpose (i<->j), C = 128 for z, 1 for mask.
// ---------------------------------------------------------------------------
__global__ __launch_bounds__(256) void transpose_kernel(
    const float* __restrict__ in, float* __restrict__ out, int C) {
  long idx = (long)blockIdx.x * 256 + threadIdx.x;
  long c = idx % C;
  long t = idx / C;
  long i = t >> 8, j = t & 255;
  out[((j << 8) + i) * C + c] = in[idx];
}

// ---------------------------------------------------------------------------
// a = mask * sigmoid(gate) * proj, packed bf16 into channel-major [c][i][k]:
//   mode 0 (outgoing): out[c*T + t]
//   mode 1 (incoming): out[c*T + (t%256)*256 + t/256]   (t = k*256 + i)
// ---------------------------------------------------------------------------
__global__ __launch_bounds__(256) void gate_pack_kernel(
    const float* __restrict__ graw, const float* __restrict__ p,
    const float* __restrict__ maskv, __bf16* __restrict__ out, int mode) {
  long idx = (long)blockIdx.x * 256 + threadIdx.x;  // t*128 + c
  long c = idx & 127;
  long t = idx >> 7;
  float val = maskv[t] * (1.0f / (1.0f + __expf(-graw[idx]))) * p[idx];
  long tt = mode ? (((t & 255) << 8) | (t >> 8)) : t;
  out[(c << 16) + tt] = (__bf16)val;
}

// ---------------------------------------------------------------------------
// Host orchestration.
// ---------------------------------------------------------------------------
extern "C" void kernel_launch(void* const* d_in, const int* in_sizes, int n_in,
                              void* d_out, int out_size, void* d_ws, size_t ws_size,
                              hipStream_t stream) {
  const int T = 65536;  // 256*256 tokens
  const float* z0   = (const float*)d_in[0];
  const float* mask = (const float*)d_in[1];
  auto F = [&](int i) { return (const float*)d_in[i]; };

  // Workspace arenas (bytes).
  char* ws = (char*)d_ws;
  float*  W0 = (float*)(ws);
  float*  W1 = (float*)(ws + 1L * 33554432);
  float*  T0 = (float*)(ws + 2L * 33554432);
  float*  T1 = (float*)(ws + 3L * 33554432);
  float*  G  = (float*)(ws + 4L * 33554432);
  __bf16* ZL = (__bf16*)(ws + 5L * 33554432);
  __bf16* Qb = ZL + (long)T * 128;
  __bf16* Kb = Qb + (long)T * 128;
  __bf16* Vb = Kb + (long)T * 128;
  __bf16* Ob = Vb + (long)T * 128;
  __bf16* AT = Ob + (long)T * 128;
  __bf16* BT = AT + (long)T * 128;
  float*  TB = (float*)(BT + (long)T * 128);
  float*  MT = TB + 4L * T;
  __bf16* Hb = Qb;  // transition hidden [T,512] aliases Q..O (exactly 64MB)

  const float rscale = 0.17677669529663687f;  // 1/sqrt(32)

  auto ln = [&](const float* x, const float* w, const float* b, __bf16* o) {
    ln128_kernel<<<T / 8, 256, 0, stream>>>(x, w, b, o, T);
  };
  auto gemm = [&](const __bf16* A, const void* W, const float* bias,
                  const float* aux, const float* mv, const float* res,
                  void* out, int M, int N, int K, float alpha, int flags,
                  long aB, long wB, int outLd, int outStride, int batch) {
    dim3 grid(M / 128, N / 16, batch);
    gemm_epi_kernel<<<grid, 256, 0, stream>>>(A, W, bias, aux, mv, res, out,
                                              M, N, K, alpha, flags, aB, wB,
                                              outLd, outStride);
  };

  auto att_stage = [&](int pb, const float* zin, const float* mv, float* zout) {
    const float *ln_w = F(pb + 0), *ln_b = F(pb + 1), *q_w = F(pb + 2),
                *k_w = F(pb + 3), *v_w = F(pb + 4), *b_w = F(pb + 5),
                *g_w = F(pb + 6), *g_b = F(pb + 7), *o_w = F(pb + 8),
                *o_b = F(pb + 9);
    ln(zin, ln_w, ln_b, ZL);
    gemm(ZL, q_w, nullptr, nullptr, nullptr, nullptr, Qb, T, 128, 128, rscale,
         F_W_F32 | F_OUT_BF16, 0, 0, 128, 1, 1);
    gemm(ZL, k_w, nullptr, nullptr, nullptr, nullptr, Kb, T, 128, 128, 1.f,
         F_W_F32 | F_OUT_BF16, 0, 0, 128, 1, 1);
    gemm(ZL, v_w, nullptr, nullptr, nullptr, nullptr, Vb, T, 128, 128, 1.f,
         F_W_F32 | F_OUT_BF16, 0, 0, 128, 1, 1);
    gemm(ZL, g_w, g_b, nullptr, nullptr, nullptr, G, T, 128, 128, 1.f,
         F_W_F32 | F_SIG, 0, 0, 128, 1, 1);
    bias_tb_kernel<<<T / 8, 256, 0, stream>>>(ZL, b_w, TB, T);
    attn_kernel<<<dim3(256, 4), 256, 81920, stream>>>(Qb, Kb, Vb, G, TB, mv, Ob);
    gemm(Ob, o_w, o_b, nullptr, nullptr, zin, zout, T, 128, 128, 1.f,
         F_W_F32 | F_RES, 0, 0, 128, 1, 1);
  };

  auto mul_stage = [&](int pb, const float* zin, const float* mv, float* zout,
                       int mode) {
    const float *ln_in_w = F(pb + 0), *ln_in_b = F(pb + 1), *ag_w = F(pb + 2),
                *ag_b = F(pb + 3), *ap_w = F(pb + 4), *ap_b = F(pb + 5),
                *bg_w = F(pb + 6), *bg_b = F(pb + 7), *bp_w = F(pb + 8),
                *bp_b = F(pb + 9), *g_w = F(pb + 10), *g_b = F(pb + 11),
                *ln_out_w = F(pb + 12), *ln_out_b = F(pb + 13),
                *z_w = F(pb + 14), *z_b = F(pb + 15);
    ln(zin, ln_in_w, ln_in_b, ZL);
    gemm(ZL, ag_w, ag_b, nullptr, nullptr, nullptr, T0, T, 128, 128, 1.f,
         F_W_F32, 0, 0, 128, 1, 1);
    gemm(ZL, ap_w, ap_b, nullptr, nullptr, nullptr, T1, T, 128, 128, 1.f,
         F_W_F32, 0, 0, 128, 1, 1);
    gate_pack_kernel<<<T * 128 / 256, 256, 0, stream>>>(T0, T1, mv, AT, mode);
    gemm(ZL, bg_w, bg_b, nullptr, nullptr, nullptr, T0, T, 128, 128, 1.f,
         F_W_F32, 0, 0, 128, 1, 1);
    gemm(ZL, bp_w, bp_b, nullptr, nullptr, nullptr, T1, T, 128, 128, 1.f,
         F_W_F32, 0, 0, 128, 1, 1);
    gate_pack_kernel<<<T * 128 / 256, 256, 0, stream>>>(T0, T1, mv, BT, mode);
    gemm(ZL, g_w, g_b, nullptr, nullptr, nullptr, G, T, 128, 128, 1.f,
         F_W_F32 | F_SIG, 0, 0, 128, 1, 1);
    // x[i,j,c] = sum_k AT[c,i,k] * BT[c,j,k]  — 128 batched 256x256x256 GEMMs
    // (bf16 W path -> async global->LDS staging)
    gemm(AT, BT, nullptr, nullptr, nullptr, nullptr, T0, 256, 256, 256, 1.f,
         0, 65536L, 65536L, 256, 128, 128);
    ln(T0, ln_out_w, ln_out_b, ZL);
    gemm(ZL, z_w, z_b, G, nullptr, zin, zout, T, 128, 128, 1.f,
         F_W_F32 | F_AUXMUL | F_RES, 0, 0, 128, 1, 1);
  };

  auto trans_stage = [&](int pb, const float* zin, const float* mv, float* zout) {
    const float *ln_w = F(pb + 0), *ln_b = F(pb + 1), *w1 = F(pb + 2),
                *b1 = F(pb + 3), *w2 = F(pb + 4), *b2 = F(pb + 5);
    ln(zin, ln_w, ln_b, ZL);
    gemm(ZL, w1, b1, nullptr, nullptr, nullptr, Hb, T, 512, 128, 1.f,
         F_W_F32 | F_RELU | F_OUT_BF16, 0, 0, 512, 1, 1);
    gemm(Hb, w2, b2, nullptr, mv, zin, zout, T, 128, 512, 1.f,
         F_W_F32 | F_MASKMUL | F_RES, 0, 0, 128, 1, 1);
  };

  // Input layout (setup_inputs insertion order): z, mask, then params:
  // att_start@2(10), att_end@12(10), mul_out@22(16), mul_in@38(16), trans@54(6)
  att_stage(2, z0, mask, W0);

  transpose_kernel<<<T * 128 / 256, 256, 0, stream>>>(W0, T0, 128);
  transpose_kernel<<<T / 256, 256, 0, stream>>>(mask, MT, 1);
  att_stage(12, T0, MT, W1);
  transpose_kernel<<<T * 128 / 256, 256, 0, stream>>>(W1, W0, 128);

  mul_stage(22, W0, mask, W1, 0);
  mul_stage(38, W1, mask, W0, 1);
  trans_stage(54, W0, mask, (float*)d_out);
}
